// EncoderMoe_30056181137729
// MI455X (gfx1250) — matrix operand order repaired and checked by reading, not compiled, not run on hardware
//
#include <hip/hip_runtime.h>
#include <hip/hip_bf16.h>

// ---------------------------------------------------------------------------
// Types / helpers
// ---------------------------------------------------------------------------
typedef __attribute__((ext_vector_type(16))) __bf16 v16bf;
typedef __attribute__((ext_vector_type(8)))  __bf16 v8bf;
typedef __attribute__((ext_vector_type(8)))  float  v8f;

__device__ __forceinline__ __bf16 f2bf(float f) {
  union { float f; unsigned u; } a; a.f = f;
  unsigned r = a.u + 0x7FFFu + ((a.u >> 16) & 1u);   // round-to-nearest-even
  union { unsigned short s; __bf16 b; } c;
  c.s = (unsigned short)(r >> 16);
  return c.b;
}

__device__ __forceinline__ float gelu_tanh(float x) {
  float x3 = x * x * x;
  return 0.5f * x * (1.0f + tanhf(0.7978845608028654f * (x + 0.044715f * x3)));
}

// ---------------------------------------------------------------------------
// WMMA bf16 GEMM:  C[z] = act( A[z] @ Wt[z]^T + bias[z] ) (+ addto[z])
//   A  : bf16 [M, K] row-major
//   Wt : bf16 [N, K] row-major (pre-transposed weight)
//   C  : fp32 or bf16 [M, N] (out_bf flag)
// Block = 128 threads = 4 waves along N; wave register tile = 32 x 64
// (2 M-frags x 4 N-frags = 8 WMMAs per k-step, 12 b128 loads per k-step).
// Block tile = 32 x 256.  Requires M % 32 == 0, N % 256 == 0, K % 32 == 0.
// A frag: lane m = lane&15, kh = lane>>4; elem j -> K = 16kh+j
//         (two contiguous 16B runs).
// B frag: lane n = lane&15, kh = lane>>4; elem j -> K = 16kh+j
//         (one contiguous 32B run of Wt row n).
// ---------------------------------------------------------------------------
__global__ __launch_bounds__(128)
void wmma_gemm_bf(const __bf16* __restrict__ A, const __bf16* __restrict__ Wt,
                  const float* __restrict__ bias, const float* __restrict__ addto,
                  void* __restrict__ Cout, int M, int N, int Kd,
                  long aZ, long wZ, long bZ, long addZ, long cZ,
                  int act, int out_bf)
{
  int z = blockIdx.z;
  A  += (size_t)z * aZ;
  Wt += (size_t)z * wZ;
  const float* biasp = bias ? bias + (size_t)z * bZ : nullptr;
  const float* addp  = addto ? addto + (size_t)z * addZ : nullptr;

  int wave = threadIdx.x >> 5;
  int lane = threadIdx.x & 31;
  int row0 = blockIdx.y * 32;
  int col0 = blockIdx.x * 256 + wave * 64;
  int m  = lane & 15;
  int kh = lane >> 4;
  int n  = lane & 15;

  v8f acc[2][4] = {};
  const __bf16* arow0 = A + (size_t)(row0 + m) * Kd + 16 * kh;
  const __bf16* arow1 = arow0 + (size_t)16 * Kd;
  const __bf16* wr0 = Wt + (size_t)(col0 + n) * Kd + 16 * kh;
  const __bf16* wr1 = wr0 + (size_t)16 * Kd;
  const __bf16* wr2 = wr0 + (size_t)32 * Kd;
  const __bf16* wr3 = wr0 + (size_t)48 * Kd;

  for (int kk = 0; kk < Kd; kk += 32) {
    v8bf a0l = *(const v8bf*)(arow0 + kk);
    v8bf a0h = *(const v8bf*)(arow0 + kk + 8);
    v8bf a1l = *(const v8bf*)(arow1 + kk);
    v8bf a1h = *(const v8bf*)(arow1 + kk + 8);
    v16bf a0 = __builtin_shufflevector(a0l, a0h, 0, 1, 2, 3, 4, 5, 6, 7,
                                       8, 9, 10, 11, 12, 13, 14, 15);
    v16bf a1 = __builtin_shufflevector(a1l, a1h, 0, 1, 2, 3, 4, 5, 6, 7,
                                       8, 9, 10, 11, 12, 13, 14, 15);
    v16bf b0 = *(const v16bf*)(wr0 + kk);
    v16bf b1 = *(const v16bf*)(wr1 + kk);
    v16bf b2 = *(const v16bf*)(wr2 + kk);
    v16bf b3 = *(const v16bf*)(wr3 + kk);
    acc[0][0] = __builtin_amdgcn_wmma_f32_16x16x32_bf16(false, a0, false, b0, (short)0, acc[0][0], false, false);
    acc[0][1] = __builtin_amdgcn_wmma_f32_16x16x32_bf16(false, a0, false, b1, (short)0, acc[0][1], false, false);
    acc[0][2] = __builtin_amdgcn_wmma_f32_16x16x32_bf16(false, a0, false, b2, (short)0, acc[0][2], false, false);
    acc[0][3] = __builtin_amdgcn_wmma_f32_16x16x32_bf16(false, a0, false, b3, (short)0, acc[0][3], false, false);
    acc[1][0] = __builtin_amdgcn_wmma_f32_16x16x32_bf16(false, a1, false, b0, (short)0, acc[1][0], false, false);
    acc[1][1] = __builtin_amdgcn_wmma_f32_16x16x32_bf16(false, a1, false, b1, (short)0, acc[1][1], false, false);
    acc[1][2] = __builtin_amdgcn_wmma_f32_16x16x32_bf16(false, a1, false, b2, (short)0, acc[1][2], false, false);
    acc[1][3] = __builtin_amdgcn_wmma_f32_16x16x32_bf16(false, a1, false, b3, (short)0, acc[1][3], false, false);
  }

  int mh = lane >> 4;
#pragma unroll
  for (int mi2 = 0; mi2 < 2; ++mi2) {
#pragma unroll
    for (int j = 0; j < 4; ++j) {
#pragma unroll
      for (int i = 0; i < 8; ++i) {
        int r = row0 + 16 * mi2 + i + 8 * mh;
        int c = col0 + 16 * j + n;
        float v = acc[mi2][j][i];
        if (biasp) v += biasp[c];
        if (act)   v = gelu_tanh(v);
        if (addp)  v += addp[(size_t)r * N + c];
        if (out_bf) ((__bf16*)Cout)[(size_t)z * cZ + (size_t)r * N + c] = f2bf(v);
        else        ((float*)Cout)[(size_t)z * cZ + (size_t)r * N + c]  = v;
      }
    }
  }
}

// ---------------------------------------------------------------------------
// Transpose + convert: W fp32 [K,N] -> Wt bf16 [N,K]  (K,N multiples of 32)
// ---------------------------------------------------------------------------
__global__ __launch_bounds__(256)
void tconv(const float* __restrict__ W, __bf16* __restrict__ Wt,
           int K, int N, long wZ, long tZ)
{
  __shared__ float tile[32][33];
  int z = blockIdx.z;
  W  += (size_t)z * wZ;
  Wt += (size_t)z * tZ;
  int n0 = blockIdx.x * 32;
  int k0 = blockIdx.y * 32;
  int tx = threadIdx.x & 31;
  int ty = threadIdx.x >> 5;   // 0..7
  for (int i = ty; i < 32; i += 8)
    tile[i][tx] = W[(size_t)(k0 + i) * N + n0 + tx];
  __syncthreads();
  for (int i = ty; i < 32; i += 8)
    Wt[(size_t)(n0 + i) * K + k0 + tx] = f2bf(tile[tx][i]);
}

// ---------------------------------------------------------------------------
// Attention scores (bf16 q,k): scores[bh,q,k] = 0.125 * (q . k) over 64.
// ---------------------------------------------------------------------------
__global__ __launch_bounds__(256)
void attn_scores(const __bf16* __restrict__ q, const __bf16* __restrict__ k,
                 float* __restrict__ scores, int S, int D, int H)
{
  int bh = blockIdx.z;
  int b = bh / H, h = bh % H;
  int wave = threadIdx.x >> 5, lane = threadIdx.x & 31;
  int wm = wave >> 2, wn = wave & 3;
  int row0 = blockIdx.y * 32 + wm * 16;
  int col0 = blockIdx.x * 64 + wn * 16;
  int m = lane & 15, kh = lane >> 4;
  int row = row0 + m;
  int col = col0 + (lane & 15);

  v8f acc = {};
  for (int kk = 0; kk < 64; kk += 32) {
    v16bf af = {};
    if (row < S) {
      const __bf16* qp = q + ((size_t)(b * S + row) * D) + h * 64 + kk + 16 * kh;
      v8bf a0 = *(const v8bf*)qp;
      v8bf a1 = *(const v8bf*)(qp + 8);
      af = __builtin_shufflevector(a0, a1, 0, 1, 2, 3, 4, 5, 6, 7,
                                   8, 9, 10, 11, 12, 13, 14, 15);
    }
    v16bf bf = {};
    if (col < S)
      bf = *(const v16bf*)(k + ((size_t)(b * S + col) * D) + h * 64 + kk + 16 * kh);
    acc = __builtin_amdgcn_wmma_f32_16x16x32_bf16(false, af, false, bf,
                                                  (short)0, acc, false, false);
  }

  int n = lane & 15, mh = lane >> 4;
#pragma unroll
  for (int i = 0; i < 8; ++i) {
    int r = row0 + i + 8 * mh;
    int c = col0 + n;
    if (r < S && c < S)
      scores[((size_t)bh * S + r) * S + c] = acc[i] * 0.125f;
  }
}

// ---------------------------------------------------------------------------
// Row softmax: fp32 scores [row, S] -> bf16 probs [row, Sp] (zero padded).
// ---------------------------------------------------------------------------
__global__ __launch_bounds__(256)
void softmax_rows_bf(const float* __restrict__ s, __bf16* __restrict__ o,
                     int S, int Sp)
{
  __shared__ float red[256];
  size_t row = blockIdx.x;
  const float* p = s + row * (size_t)S;
  __bf16* op = o + row * (size_t)Sp;
  int t = threadIdx.x;

  float v = (t < S) ? p[t] : -3.4e38f;
  red[t] = v;
  __syncthreads();
  for (int off = 128; off > 0; off >>= 1) {
    if (t < off) red[t] = fmaxf(red[t], red[t + off]);
    __syncthreads();
  }
  float mx = red[0];
  __syncthreads();
  float e = (t < S) ? __expf(v - mx) : 0.0f;
  red[t] = e;
  __syncthreads();
  for (int off = 128; off > 0; off >>= 1) {
    if (t < off) red[t] += red[t + off];
    __syncthreads();
  }
  float inv = 1.0f / red[0];
  if (t < Sp) op[t] = f2bf((t < S) ? e * inv : 0.0f);
}

// ---------------------------------------------------------------------------
// out_bf[b,q,h,:] = att_bf[bh,q,:Sp] @ v[b,:,h,:]  (K padded to Sp, B guarded)
// ---------------------------------------------------------------------------
__global__ __launch_bounds__(256)
void attn_av(const __bf16* __restrict__ att, const float* __restrict__ v,
             __bf16* __restrict__ out, int S, int Sp, int D, int H)
{
  int bh = blockIdx.z;
  int b = bh / H, h = bh % H;
  int wave = threadIdx.x >> 5, lane = threadIdx.x & 31;
  int wm = wave >> 2, wn = wave & 3;
  int row0 = blockIdx.y * 32 + wm * 16;
  int col0 = wn * 16;  // N = 64, gridDim.x == 1
  int m = lane & 15, kh = lane >> 4;
  int row = row0 + m;
  int n = lane & 15;

  const __bf16* ap = att + ((size_t)bh * S + (row < S ? row : 0)) * Sp;
  v8f acc = {};
  for (int kk = 0; kk < Sp; kk += 32) {
    v16bf af = {};
    if (row < S) {
      v8bf a0 = *(const v8bf*)(ap + kk + 16 * kh);
      v8bf a1 = *(const v8bf*)(ap + kk + 8 + 16 * kh);
      af = __builtin_shufflevector(a0, a1, 0, 1, 2, 3, 4, 5, 6, 7,
                                   8, 9, 10, 11, 12, 13, 14, 15);
    }
    v16bf bf;
#pragma unroll
    for (int j = 0; j < 16; ++j) {
      int K = kk + 16 * kh + j;
      bf[j] = f2bf((K < S) ? v[((size_t)(b * S + K)) * D + h * 64 + col0 + n] : 0.0f);
    }
    acc = __builtin_amdgcn_wmma_f32_16x16x32_bf16(false, af, false, bf,
                                                  (short)0, acc, false, false);
  }

  int mh = lane >> 4;
#pragma unroll
  for (int i = 0; i < 8; ++i) {
    int r = row0 + i + 8 * mh;
    if (r < S)
      out[(size_t)(b * S + r) * D + h * 64 + col0 + n] = f2bf(acc[i]);
  }
}

// ---------------------------------------------------------------------------
// LayerNorm over D = 768; writes fp32 and (optionally) bf16.
// ---------------------------------------------------------------------------
__global__ __launch_bounds__(256)
void layernorm768(const float* __restrict__ x, float* __restrict__ o,
                  __bf16* __restrict__ obf,
                  const float* __restrict__ s, const float* __restrict__ b)
{
  __shared__ float r1[256];
  __shared__ float r2[256];
  size_t row = blockIdx.x;
  const float* xp = x + row * 768;
  int t = threadIdx.x;
  float a0 = xp[t], a1 = xp[t + 256], a2 = xp[t + 512];
  r1[t] = a0 + a1 + a2;
  r2[t] = a0 * a0 + a1 * a1 + a2 * a2;
  __syncthreads();
  for (int off = 128; off > 0; off >>= 1) {
    if (t < off) { r1[t] += r1[t + off]; r2[t] += r2[t + off]; }
    __syncthreads();
  }
  float mu = r1[0] * (1.0f / 768.0f);
  float var = r2[0] * (1.0f / 768.0f) - mu * mu;
  float rs = rsqrtf(var + 1e-6f);
  float y0 = (a0 - mu) * rs * s[t]       + b[t];
  float y1 = (a1 - mu) * rs * s[t + 256] + b[t + 256];
  float y2 = (a2 - mu) * rs * s[t + 512] + b[t + 512];
  float* op = o + row * 768;
  op[t] = y0; op[t + 256] = y1; op[t + 512] = y2;
  if (obf) {
    __bf16* bp = obf + row * 768;
    bp[t] = f2bf(y0); bp[t + 256] = f2bf(y1); bp[t + 512] = f2bf(y2);
  }
}

// ---------------------------------------------------------------------------
// Misc elementwise kernels
// ---------------------------------------------------------------------------
__global__ void add_posemb(const float* __restrict__ x, const float* __restrict__ p,
                           float* __restrict__ o, int SD, long total)
{
  long i = (long)blockIdx.x * blockDim.x + threadIdx.x;
  if (i < total) o[i] = x[i] + p[i % SD];
}

__global__ void zero_f(float* __restrict__ p, long n)
{
  long i = (long)blockIdx.x * blockDim.x + threadIdx.x;
  if (i < n) p[i] = 0.0f;
}

// ---------------------------------------------------------------------------
// Router: logits = h @ rw (D=768, E=8), softmax, top-2 (ties -> lower index)
// ---------------------------------------------------------------------------
__global__ void router_topk(const float* __restrict__ h, const float* __restrict__ rw,
                            float* __restrict__ topv, int* __restrict__ topi, int T)
{
  int t = blockIdx.x * blockDim.x + threadIdx.x;
  if (t >= T) return;
  float lg[8] = {0, 0, 0, 0, 0, 0, 0, 0};
  const float* hp = h + (size_t)t * 768;
  for (int d = 0; d < 768; ++d) {
    float x = hp[d];
    const float* w = rw + (size_t)d * 8;
#pragma unroll
    for (int e = 0; e < 8; ++e) lg[e] += x * w[e];
  }
  float mx = lg[0];
#pragma unroll
  for (int e = 1; e < 8; ++e) mx = fmaxf(mx, lg[e]);
  float se = 0.0f;
#pragma unroll
  for (int e = 0; e < 8; ++e) { lg[e] = __expf(lg[e] - mx); se += lg[e]; }
  float inv = 1.0f / se;
#pragma unroll
  for (int e = 0; e < 8; ++e) lg[e] *= inv;
  int i0 = 0;
#pragma unroll
  for (int e = 1; e < 8; ++e) if (lg[e] > lg[i0]) i0 = e;
  int i1 = -1;
#pragma unroll
  for (int e = 0; e < 8; ++e)
    if (e != i0 && (i1 < 0 || lg[e] > lg[i1])) i1 = e;
  topi[t * 2]     = i0;
  topi[t * 2 + 1] = i1;
  topv[t * 2]     = lg[i0];
  topv[t * 2 + 1] = lg[i1];
}

// ---------------------------------------------------------------------------
// Capacity dispatch, slot-major priority (all top-1 before top-2), per group.
// ---------------------------------------------------------------------------
__global__ void moe_dispatch(const int* __restrict__ topi, int* __restrict__ dest, int Ng)
{
  int n = blockIdx.x * blockDim.x + threadIdx.x;
  if (n >= Ng) return;
  int cnt[8] = {0, 0, 0, 0, 0, 0, 0, 0};
  for (int k = 0; k < 2; ++k) {
    for (int g = 0; g < 392; ++g) {
      int t = n * 392 + g;
      int e = topi[t * 2 + k];
      int d = -1;
      if (cnt[e] < 98) { d = (e * Ng + n) * 98 + cnt[e]; cnt[e]++; }
      dest[t * 2 + k] = d;
    }
  }
}

__global__ __launch_bounds__(256)
void moe_gather(const float* __restrict__ h, const int* __restrict__ dest,
                __bf16* __restrict__ buf)
{
  int idx = blockIdx.x;  // t*2 + k
  int d = dest[idx];
  if (d < 0) return;
  int t = idx >> 1;
  const float* hp = h + (size_t)t * 768;
  __bf16* bp = buf + (size_t)d * 768;
  for (int j = threadIdx.x; j < 768; j += 256) bp[j] = f2bf(hp[j]);
}

__global__ __launch_bounds__(256)
void moe_combine(const float* __restrict__ y, const int* __restrict__ dest,
                 const float* __restrict__ topv, float* __restrict__ x)
{
  int t = blockIdx.x;
  int d0 = dest[t * 2], d1 = dest[t * 2 + 1];
  float g0 = topv[t * 2], g1 = topv[t * 2 + 1];
  float* xp = x + (size_t)t * 768;
  for (int j = threadIdx.x; j < 768; j += 256) {
    float a = 0.0f;
    if (d0 >= 0) a += g0 * y[(size_t)d0 * 768 + j];
    if (d1 >= 0) a += g1 * y[(size_t)d1 * 768 + j];
    xp[j] += a;
  }
}

// ---------------------------------------------------------------------------
// Host orchestration
// ---------------------------------------------------------------------------
static inline void launch_gemm(hipStream_t stream, const __bf16* A, const __bf16* Wt,
                               const float* bias, const float* addto, void* C,
                               int M, int N, int K, int act, int out_bf, int Z,
                               long aZ, long wZ, long bZ, long addZ, long cZ)
{
  dim3 grid(N / 256, M / 32, Z);
  wmma_gemm_bf<<<grid, dim3(128), 0, stream>>>(A, Wt, bias, addto, C, M, N, K,
                                               aZ, wZ, bZ, addZ, cZ, act, out_bf);
}

static inline void launch_tconv(hipStream_t stream, const float* W, __bf16* Wt,
                                int K, int N, int Z, long wZ, long tZ)
{
  dim3 grid(N / 32, K / 32, Z);
  tconv<<<grid, dim3(256), 0, stream>>>(W, Wt, K, N, wZ, tZ);
}

extern "C" void kernel_launch(void* const* d_in, const int* in_sizes, int n_in,
                              void* d_out, int out_size, void* d_ws, size_t ws_size,
                              hipStream_t stream)
{
  (void)in_sizes; (void)n_in; (void)out_size; (void)ws_size;

  const float* in_x     = (const float*)d_in[0];
  const float* posemb   = (const float*)d_in[1];
  const float* ln1_s    = (const float*)d_in[2];
  const float* ln1_b    = (const float*)d_in[3];
  const float* wq       = (const float*)d_in[4];
  const float* bq       = (const float*)d_in[5];
  const float* wk       = (const float*)d_in[6];
  const float* bk       = (const float*)d_in[7];
  const float* wv       = (const float*)d_in[8];
  const float* bv       = (const float*)d_in[9];
  const float* wo       = (const float*)d_in[10];
  const float* bo       = (const float*)d_in[11];
  const float* ln2_s    = (const float*)d_in[12];
  const float* ln2_b    = (const float*)d_in[13];
  const float* dw1      = (const float*)d_in[14];
  const float* db1      = (const float*)d_in[15];
  const float* dw2      = (const float*)d_in[16];
  const float* db2      = (const float*)d_in[17];
  const float* router_w = (const float*)d_in[18];
  const float* mw1      = (const float*)d_in[19];
  const float* mb1      = (const float*)d_in[20];
  const float* mw2      = (const float*)d_in[21];
  const float* mb2      = (const float*)d_in[22];
  const float* out_s    = (const float*)d_in[23];
  const float* out_b    = (const float*)d_in[24];

  const int D = 768, H = 12, M = 3072, S = 196, Sp = 224, B = 32;
  const int E = 8, Ng = 16, CAP = 98;
  const int TOK = B * S;                 // 6272 (multiple of 32)
  const size_t TD = (size_t)TOK * D;     // 4,816,896
  const int SLOTS = E * Ng * CAP;        // 12544 (= 2*TOK, multiple of 32)
  const int ER = Ng * CAP;               // 1568 rows per expert

  // ---------------- workspace layout (in floats) ----------------
  float* ws = (float*)d_ws;
  float* xb = ws;                         float* p = xb + TD;       // residual
  float* hb = p;                          p += TD;                  // LN out fp32
  float* vb = p;                          p += TD;                  // V fp32
  __bf16* hb_bf  = (__bf16*)p;            p += TD / 2;              // LN out bf16
  __bf16* aob_bf = (__bf16*)p;            p += TD / 2;              // attn out bf16
  __bf16* qk_bf  = (__bf16*)p;            p += TD;                  // q|k bf16  / MoE in buf
  float*  r1     = p;                     p += (size_t)SLOTS * M / 2; // scores | moeh_bf
  float*  r2     = p;                     p += (size_t)SLOTS * D;   // att_bf | h1_bf | ybuf
  __bf16* wt     = (__bf16*)p;            p += (size_t)E * D * M / 2; // weight^T bf16
  float*  topv   = p;                     p += 2 * TOK;
  int*    topi   = (int*)p;               p += 2 * TOK;
  int*    dest   = (int*)p;

  __bf16* qb_bf  = qk_bf;
  __bf16* kb_bf  = qk_bf + TD;
  __bf16* buf_bf = qk_bf;                 // MoE expert input (bf16), 2*TD elems
  float*  scores = r1;                    // [B*H*S, S] fp32
  __bf16* moeh_bf = (__bf16*)r1;          // [SLOTS, M] bf16
  __bf16* att_bf = (__bf16*)r2;           // [B*H*S, Sp] bf16
  __bf16* h1_bf  = (__bf16*)r2;           // [TOK, M] bf16
  float*  ybuf   = r2;                    // [SLOTS, D] fp32

  // x = x + posemb
  add_posemb<<<(int)((TD + 255) / 256), 256, 0, stream>>>(in_x, posemb, xb,
                                                          S * D, (long)TD);

  int di = 0, mi = 0;
  for (int l = 0; l < 12; ++l) {
    const bool is_moe = (l & 1);  // MOE_LAYERS = 1,3,5,7,9,11
    const size_t DD = (size_t)D * D;

    // h = LN1(x)
    layernorm768<<<TOK, 256, 0, stream>>>(xb, hb, hb_bf,
                                          ln1_s + (size_t)l * D, ln1_b + (size_t)l * D);
    // q, k (bf16 out), v (fp32 out)
    launch_tconv(stream, wq + l * DD, wt, D, D, 1, 0, 0);
    launch_gemm(stream, hb_bf, wt, bq + (size_t)l * D, nullptr, qb_bf,
                TOK, D, D, 0, 1, 1, 0, 0, 0, 0, 0);
    launch_tconv(stream, wk + l * DD, wt, D, D, 1, 0, 0);
    launch_gemm(stream, hb_bf, wt, bk + (size_t)l * D, nullptr, kb_bf,
                TOK, D, D, 0, 1, 1, 0, 0, 0, 0, 0);
    launch_tconv(stream, wv + l * DD, wt, D, D, 1, 0, 0);
    launch_gemm(stream, hb_bf, wt, bv + (size_t)l * D, nullptr, vb,
                TOK, D, D, 0, 0, 1, 0, 0, 0, 0, 0);
    // attention
    attn_scores<<<dim3(4, 7, B * H), 256, 0, stream>>>(qb_bf, kb_bf, scores, S, D, H);
    softmax_rows_bf<<<B * H * S, 256, 0, stream>>>(scores, att_bf, S, Sp);
    attn_av<<<dim3(1, 7, B * H), 256, 0, stream>>>(att_bf, vb, aob_bf, S, Sp, D, H);
    // x = x + ao @ wo + bo
    launch_tconv(stream, wo + l * DD, wt, D, D, 1, 0, 0);
    launch_gemm(stream, aob_bf, wt, bo + (size_t)l * D, xb, xb,
                TOK, D, D, 0, 0, 1, 0, 0, 0, 0, 0);
    // h = LN2(x)
    layernorm768<<<TOK, 256, 0, stream>>>(xb, hb, hb_bf,
                                          ln2_s + (size_t)l * D, ln2_b + (size_t)l * D);

    if (is_moe) {
      router_topk<<<(TOK + 255) / 256, 256, 0, stream>>>(
          hb, router_w + (size_t)mi * D * E, topv, topi, TOK);
      moe_dispatch<<<1, 16, 0, stream>>>(topi, dest, Ng);
      zero_f<<<(int)((TD + 255) / 256), 256, 0, stream>>>((float*)buf_bf, (long)TD);
      moe_gather<<<2 * TOK, 256, 0, stream>>>(hb, dest, buf_bf);
      // per-expert: moeh = gelu(buf @ w1 + b1)   [bf16 out]
      launch_tconv(stream, mw1 + (size_t)mi * E * D * M, wt, D, M, E,
                   (long)D * M, (long)D * M);
      launch_gemm(stream, buf_bf, wt, mb1 + (size_t)mi * E * M, nullptr, moeh_bf,
                  ER, M, D, /*gelu*/1, /*bf16*/1, E,
                  (long)ER * D, (long)D * M, (long)M, 0, (long)ER * M);
      // per-expert: y = moeh @ w2 + b2   [fp32 out]
      launch_tconv(stream, mw2 + (size_t)mi * E * M * D, wt, M, D, E,
                   (long)M * D, (long)M * D);
      launch_gemm(stream, moeh_bf, wt, mb2 + (size_t)mi * E * D, nullptr, ybuf,
                  ER, D, M, 0, 0, E,
                  (long)ER * M, (long)M * D, (long)D, 0, (long)ER * D);
      moe_combine<<<TOK, 256, 0, stream>>>(ybuf, dest, topv, xb);
      mi++;
    } else {
      // h1 = gelu(h @ dw1 + db1) [bf16]; x = x + h1 @ dw2 + db2
      launch_tconv(stream, dw1 + (size_t)di * D * M, wt, D, M, 1, 0, 0);
      launch_gemm(stream, hb_bf, wt, db1 + (size_t)di * M, nullptr, h1_bf,
                  TOK, M, D, /*gelu*/1, /*bf16*/1, 1, 0, 0, 0, 0, 0);
      launch_tconv(stream, dw2 + (size_t)di * M * D, wt, M, D, 1, 0, 0);
      launch_gemm(stream, h1_bf, wt, db2 + (size_t)di * D, xb, xb,
                  TOK, D, M, 0, 0, 1, 0, 0, 0, 0, 0);
      di++;
    }
  }

  // final LN -> d_out (fp32)
  layernorm768<<<TOK, 256, 0, stream>>>(xb, (float*)d_out, nullptr, out_s, out_b);
}
